// CrossAttention_59004260712847
// MI455X (gfx1250) — matrix operand order repaired and checked by reading, not compiled, not run on hardware
//
#include <hip/hip_runtime.h>

// CDNA5 / gfx1250 cross-attention.
// Roofline: ~120 GFLOP vs ~110MB unique bytes (fits in 192MB L2) -> compute
// bound -> run everything on V_WMMA_F32_16X16X32_BF16 (bf16 in, f32 accum).
// K/Q tile staging uses GLOBAL_LOAD_ASYNC_TO_LDS (ASYNCcnt), double-buffered
// so async DMA overlaps the WMMA stream.

typedef __attribute__((ext_vector_type(16))) __bf16 v16bf;
typedef __attribute__((ext_vector_type(8)))  float  v8f;

union PairU { unsigned u; unsigned short s[2]; };
union AFrag { v16bf v; unsigned u[8]; };
union BfU   { __bf16 h; unsigned short s; };

// fp32 -> bf16 via native fptrunc (gfx1250 has v_cvt_pk_bf16_f32, RNE).
__device__ __forceinline__ unsigned short cvt_bf16(float f) {
  BfU t; t.h = (__bf16)f; return t.s;
}

__device__ __forceinline__ unsigned ldsPair(const unsigned short* p) {
  return *(const unsigned*)(p);   // 4B-aligned by construction -> ds_load_b32
}

__device__ __forceinline__ v8f wmma_bf16(const AFrag& a, const AFrag& b, v8f c) {
  return __builtin_amdgcn_wmma_f32_16x16x32_bf16(
      false, a.v, false, b.v, (short)0, c, false, false);
}

// ---- gfx1250 async global->LDS copy (ASYNCcnt) with safe fallback ----------
#if __has_builtin(__builtin_amdgcn_global_load_async_to_lds_b32)
#define HAVE_ASYNC_LDS 1
#else
#define HAVE_ASYNC_LDS 0
#endif

typedef __attribute__((address_space(1))) int gint_t;
typedef __attribute__((address_space(3))) int lint_t;

__device__ __forceinline__ void g2l_async_b32(void* lds, const void* g) {
#if HAVE_ASYNC_LDS
  // AS1 pointer = full 64-bit VA; AS3 pointer = low 32 bits of generic LDS VA
  // (ISA 10.2: only the low 32 address bits are used). Builtin prototype takes int* args.
  __builtin_amdgcn_global_load_async_to_lds_b32(
      (gint_t*)(unsigned long long)g,
      (lint_t*)(unsigned)(unsigned long long)lds,
      0, 0);
#else
  *(unsigned*)lds = *(const unsigned*)g;
#endif
}

__device__ __forceinline__ void wait_async_all() {
#if HAVE_ASYNC_LDS
#if __has_builtin(__builtin_amdgcn_s_wait_asynccnt)
  __builtin_amdgcn_s_wait_asynccnt(0);
#else
  asm volatile("s_wait_asynccnt 0" ::: "memory");
#endif
#endif
}

// ---------------------------------------------------------------------------
// GEMM: C(bf16) = cscale * (A(f32, MxK row-major) * B(f32, KxN row-major))
// Block tile 128x128, K-step 32. 8 waves; wave (wm,wn) owns 64x32 (4x2 tiles).
// fp32 loads must be converted, so these stay on the VGPR path + prefetch.
// ---------------------------------------------------------------------------
#define LDT 34  // padded LDS stride (elements) for 32-wide K tiles

__global__ __launch_bounds__(256)
void gemm_f32_bf16(const float* __restrict__ A, const float* __restrict__ B,
                   unsigned short* __restrict__ C, int N, int K, float cscale) {
  __shared__ unsigned short lds_a[128 * LDT];   // [row][k]
  __shared__ unsigned short lds_bt[128 * LDT];  // [n][k]  (B transposed)

  const int tid = threadIdx.x;
  const int w = tid >> 5, lane = tid & 31;
  const int lr = lane & 15, lh = lane >> 4;
  const int wm = (w >> 2) * 64, wn = (w & 3) * 32;
  const long m0 = (long)blockIdx.y * 128;
  const long n0 = (long)blockIdx.x * 128;

  v8f acc[4][2];
#pragma unroll
  for (int i = 0; i < 4; ++i)
#pragma unroll
    for (int j = 0; j < 2; ++j)
      acc[i][j] = (v8f){0.f, 0.f, 0.f, 0.f, 0.f, 0.f, 0.f, 0.f};

  for (int k0 = 0; k0 < K; k0 += 32) {
    __syncthreads();
    // A tile: 128 rows x 32 k, coalesced (one row per wave-read), cvt->bf16
#pragma unroll
    for (int it = 0; it < 16; ++it) {
      int idx = tid + it * 256;
      int r = idx >> 5, kk = idx & 31;
      lds_a[r * LDT + kk] = cvt_bf16(A[(m0 + r) * (long)K + k0 + kk]);
    }
    // B tile transposed: rows k0..k0+31, cols n0..n0+127, coalesced along n
#pragma unroll
    for (int it = 0; it < 16; ++it) {
      int idx = tid + it * 256;
      int nn = idx & 127, kk = idx >> 7;
      lds_bt[nn * LDT + kk] = cvt_bf16(B[(long)(k0 + kk) * N + n0 + nn]);
    }
    // Prefetch next K-step into L2/L0 while we compute this one.
    if (k0 + 32 < K) {
      __builtin_prefetch(&A[(m0 + (tid >> 5)) * (long)K + k0 + 32 + (tid & 31)], 0, 0);
      __builtin_prefetch(&B[(long)(k0 + 32 + (tid >> 7)) * N + n0 + (tid & 127)], 0, 0);
    }
    __syncthreads();

    // Build fragments: contiguous k order (k = 16*lh + s), same for A and B.
    AFrag a[4], b[2];
#pragma unroll
    for (int mi = 0; mi < 4; ++mi)
#pragma unroll
      for (int i = 0; i < 8; ++i) {
        int kk = lh * 16 + 2 * i;  // A: row-in-lane
        a[mi].u[i] = ldsPair(&lds_a[(wm + mi * 16 + lr) * LDT + kk]);
      }
#pragma unroll
    for (int ni = 0; ni < 2; ++ni)
#pragma unroll
      for (int i = 0; i < 8; ++i) {
        int kk = lh * 16 + 2 * i;                        // B: K-pairs per VGPR
        b[ni].u[i] = ldsPair(&lds_bt[(wn + ni * 16 + lr) * LDT + kk]);
      }
#pragma unroll
    for (int mi = 0; mi < 4; ++mi)
#pragma unroll
      for (int ni = 0; ni < 2; ++ni)
        acc[mi][ni] = wmma_bf16(a[mi], b[ni], acc[mi][ni]);
  }

  // C/D layout: VGPR r -> row r + 8*half, col = lane&15
#pragma unroll
  for (int mi = 0; mi < 4; ++mi)
#pragma unroll
    for (int ni = 0; ni < 2; ++ni)
#pragma unroll
      for (int r = 0; r < 8; ++r) {
        long row = m0 + wm + mi * 16 + r + 8 * lh;
        long col = n0 + wn + ni * 16 + lr;
        C[row * N + col] = cvt_bf16(acc[mi][ni][r] * cscale);
      }
}

// ---------------------------------------------------------------------------
// Flash attention: one block = 128 queries of one (b,h); 8 waves x 16 rows.
// Double-buffered 64-key tiles (async K + transposing V), online softmax,
// P->LDS round trip for the C-layout -> A-layout change.
// ---------------------------------------------------------------------------
#define LQS 66  // padded stride for 64-wide d tiles

__global__ __launch_bounds__(256)
void attn_kernel(const unsigned short* __restrict__ Q,   // [8192][1024] bf16
                 const unsigned short* __restrict__ KV,  // [8192][2048] bf16
                 float* __restrict__ Out) {               // [8192][1024] f32
  __shared__ unsigned short lds_q[128 * LQS];        // [qrow][d]
  __shared__ unsigned short lds_k[2][64 * LQS];      // [key][d]   (B N-major)
  __shared__ unsigned short lds_vt[2][64 * LQS];     // [dv][key]  (V^T)
  __shared__ unsigned short lds_p[8 * 16 * LQS];     // per-wave P scratch

  const int tid = threadIdx.x;
  const int w = tid >> 5, lane = tid & 31;
  const int lr = lane & 15, lh = lane >> 4;
  const int h = blockIdx.y, b = blockIdx.z;
  const long qrow0 = (long)b * 2048 + (long)blockIdx.x * 128;
  const unsigned* qsrc = (const unsigned*)Q;    // row stride 512 uints
  const unsigned* kvsrc = (const unsigned*)KV;  // row stride 1024 uints

  // Issue one 64-key tile load (async K + transposing V) into buffer `buf`.
  auto load_tile = [&](int jt, int buf) {
    const long jrow0 = (long)b * 2048 + jt * 64;
#pragma unroll
    for (int it = 0; it < 8; ++it) {
      int idx = tid + it * 256;
      int r = idx >> 5, cu = idx & 31;
      // K tile: straight copy -> async global->LDS (ASYNCcnt path).
      g2l_async_b32(&lds_k[buf][r * LQS + 2 * cu],
                    &kvsrc[(jrow0 + r) * 1024 + h * 32 + cu]);
      // V tile: needs 16-bit transpose, so VGPR path + two b16 DS stores.
      unsigned vv = kvsrc[(jrow0 + r) * 1024 + 512 + h * 32 + cu];
      PairU t; t.u = vv;
      lds_vt[buf][(2 * cu) * LQS + r] = t.s[0];
      lds_vt[buf][(2 * cu + 1) * LQS + r] = t.s[1];
    }
  };

  // Stage Q tile (async direct-to-LDS) and the first K/V tile.
#pragma unroll
  for (int it = 0; it < 16; ++it) {
    int idx = tid + it * 256;
    int r = idx >> 5, cu = idx & 31;
    g2l_async_b32(&lds_q[r * LQS + 2 * cu], &qsrc[(qrow0 + r) * 512 + h * 32 + cu]);
  }
  load_tile(0, 0);
  wait_async_all();
  __syncthreads();

  // Per-wave Q fragments (16 rows x 64 d = two K-steps), kept in registers.
  // Q was pre-scaled by 1/sqrt(dqk) in the projection GEMM.
  AFrag aq[2];
#pragma unroll
  for (int ks = 0; ks < 2; ++ks)
#pragma unroll
    for (int i = 0; i < 8; ++i) {
      int kk = ks * 32 + lh * 16 + 2 * i;
      aq[ks].u[i] = ldsPair(&lds_q[(w * 16 + lr) * LQS + kk]);
    }

  v8f o[4];
  float m[8], l[8];
#pragma unroll
  for (int n = 0; n < 4; ++n) o[n] = (v8f){0.f,0.f,0.f,0.f,0.f,0.f,0.f,0.f};
#pragma unroll
  for (int r = 0; r < 8; ++r) { m[r] = -1e30f; l[r] = 0.f; }

  unsigned short* pp = &lds_p[w * 16 * LQS];

  for (int jt = 0; jt < 32; ++jt) {
    const int cur = jt & 1;
    if (jt > 0) wait_async_all();  // tile `jt` asyncs (issued last iteration)
    __syncthreads();               // tile data visible; buf 1-cur free to reuse
    if (jt + 1 < 32) load_tile(jt + 1, cur ^ 1);  // overlap with compute below

    // S = Q * K^T  (4 key tiles x 2 K-steps)
    v8f s[4];
#pragma unroll
    for (int n = 0; n < 4; ++n) s[n] = (v8f){0.f,0.f,0.f,0.f,0.f,0.f,0.f,0.f};
#pragma unroll
    for (int n = 0; n < 4; ++n)
#pragma unroll
      for (int ks = 0; ks < 2; ++ks) {
        AFrag bk;
#pragma unroll
        for (int i = 0; i < 8; ++i)
          bk.u[i] = ldsPair(&lds_k[cur][(n * 16 + lr) * LQS + ks * 32 + lh * 16 + 2 * i]);
        s[n] = wmma_bf16(aq[ks], bk, s[n]);
      }

    // Online softmax. Row of VGPR r = w*16 + r + 8*lh; 16-lane row reduction.
    float cf[8], rs[8];
#pragma unroll
    for (int r = 0; r < 8; ++r) {
      float rmax = -1e30f;
#pragma unroll
      for (int n = 0; n < 4; ++n) rmax = fmaxf(rmax, s[n][r]);
      rmax = fmaxf(rmax, __shfl_xor(rmax, 1));
      rmax = fmaxf(rmax, __shfl_xor(rmax, 2));
      rmax = fmaxf(rmax, __shfl_xor(rmax, 4));
      rmax = fmaxf(rmax, __shfl_xor(rmax, 8));
      float mn = fmaxf(m[r], rmax);
      cf[r] = __expf(m[r] - mn);
      m[r] = mn;
      rs[r] = 0.f;
    }
#pragma unroll
    for (int n = 0; n < 4; ++n)
#pragma unroll
      for (int r = 0; r < 8; ++r) {
        float p = __expf(s[n][r] - m[r]);
        s[n][r] = p;
        rs[r] += p;
      }
#pragma unroll
    for (int r = 0; r < 8; ++r) {
      rs[r] += __shfl_xor(rs[r], 1);
      rs[r] += __shfl_xor(rs[r], 2);
      rs[r] += __shfl_xor(rs[r], 4);
      rs[r] += __shfl_xor(rs[r], 8);
      l[r] = l[r] * cf[r] + rs[r];
    }
#pragma unroll
    for (int n = 0; n < 4; ++n)
#pragma unroll
      for (int r = 0; r < 8; ++r) o[n][r] *= cf[r];

    // P: C-layout -> LDS (bf16) -> A-layout frags (DS ops in-order per wave).
#pragma unroll
    for (int n = 0; n < 4; ++n)
#pragma unroll
      for (int r = 0; r < 8; ++r)
        pp[(r + 8 * lh) * LQS + n * 16 + lr] = cvt_bf16(s[n][r]);

    AFrag ap[2];
#pragma unroll
    for (int ks = 0; ks < 2; ++ks)
#pragma unroll
      for (int i = 0; i < 8; ++i) {
        int kk = ks * 32 + lh * 16 + 2 * i;
        ap[ks].u[i] = ldsPair(&pp[lr * LQS + kk]);
      }

    // O += P * V   (4 dv tiles x 2 key K-steps)
#pragma unroll
    for (int n = 0; n < 4; ++n)
#pragma unroll
      for (int ks = 0; ks < 2; ++ks) {
        AFrag bv;
#pragma unroll
        for (int i = 0; i < 8; ++i)
          bv.u[i] = ldsPair(&lds_vt[cur][(n * 16 + lr) * LQS + ks * 32 + lh * 16 + 2 * i]);
        o[n] = wmma_bf16(ap[ks], bv, o[n]);
      }
  }

  // Normalize and store fp32 output.
#pragma unroll
  for (int n = 0; n < 4; ++n)
#pragma unroll
    for (int r = 0; r < 8; ++r) {
      long row = qrow0 + w * 16 + r + 8 * lh;
      int col = h * 64 + n * 16 + lr;
      Out[row * 1024 + col] = o[n][r] / l[r];
    }
}

// ---------------------------------------------------------------------------
extern "C" void kernel_launch(void* const* d_in, const int* in_sizes, int n_in,
                              void* d_out, int out_size, void* d_ws, size_t ws_size,
                              hipStream_t stream) {
  const float* x   = (const float*)d_in[0];  // [4,2048,1024]
  const float* ctx = (const float*)d_in[1];  // [4,2048,1024]
  const float* Wq  = (const float*)d_in[2];  // [1024,1024]
  const float* Wkv = (const float*)d_in[3];  // [1024,2048]
  float* out = (float*)d_out;                // [4,2048,1024]

  unsigned short* qbf  = (unsigned short*)d_ws;              // 8192*1024 bf16
  unsigned short* kvbf = qbf + (size_t)8192 * 1024;          // 8192*2048 bf16

  dim3 blk(256);
  // Q projection pre-scaled by 1/sqrt(dqk) = 0.125.
  gemm_f32_bf16<<<dim3(1024 / 128, 8192 / 128), blk, 0, stream>>>(x, Wq, qbf, 1024, 1024, 0.125f);
  gemm_f32_bf16<<<dim3(2048 / 128, 8192 / 128), blk, 0, stream>>>(ctx, Wkv, kvbf, 2048, 1024, 1.0f);
  attn_kernel<<<dim3(2048 / 128, 16, 4), blk, 0, stream>>>(qbf, kvbf, out);
}
